// SOAP_predictor_model_52888227283050
// MI455X (gfx1250) — matrix operand order repaired and checked by reading, not compiled, not run on hardware
//
#include <hip/hip_runtime.h>

// ---------------- types ----------------
typedef __attribute__((ext_vector_type(16))) _Float16 v16h;
typedef __attribute__((ext_vector_type(8)))  _Float16 v8h;
typedef __attribute__((ext_vector_type(8)))  float    v8f;
typedef __attribute__((ext_vector_type(4)))  float    v4f;

#define GAS __attribute__((address_space(1)))

__device__ __forceinline__ float ldg1(const float* p) {
  return *(const GAS float*)p;
}
__device__ __forceinline__ v4f ldg4(const float* p) {
  return *(const GAS v4f*)p;
}
__device__ __forceinline__ void stg1(float* p, float v) {
  *(GAS float*)p = v;
}

__device__ __forceinline__ v8f wmma16(v16h a, v16h b, v8f c) {
  return __builtin_amdgcn_wmma_f32_16x16x32_f16(
      false, a, false, b, (short)0, c, false, false);
}

// ---------------- fragment loaders ----------------
// A fragment (16x32): lanes 0-15: elems0..7=K k0..k0+7, elems8..15=K k0+16..23
//                     lanes16-31: elems0..7=K k0+8..15, elems8..15=K k0+24..31
template <int KVALID>
__device__ __forceinline__ v16h fragA(const _Float16* base, int row0, int stride,
                                      int k0) {
  const int lane = threadIdx.x & 31;
  const int m = lane & 15, hi = lane >> 4;
  const _Float16* p = base + (row0 + m) * stride;
  const int ka = k0 + hi * 8, kb = ka + 16;
  const v8h a = *(const v8h*)(p + ka);   // always in-bounds (padded strides)
  const v8h b = *(const v8h*)(p + kb);
  v16h r;
  if (k0 + 32 <= KVALID) {               // compile-time foldable
#pragma unroll
    for (int i = 0; i < 8; ++i) { r[i] = a[i]; r[8 + i] = b[i]; }
  } else {
#pragma unroll
    for (int i = 0; i < 8; ++i) {
      r[i]     = (ka + i < KVALID) ? a[i] : (_Float16)0.f;
      r[8 + i] = (kb + i < KVALID) ? b[i] : (_Float16)0.f;
    }
  }
  return r;
}

// A fragment from global f32 (row-major, tight stride) with f16 convert.
template <int KVALID>
__device__ __forceinline__ v16h fragA(const float* base, int row0, int stride,
                                      int k0) {
  const int lane = threadIdx.x & 31;
  const int m = lane & 15, hi = lane >> 4;
  const float* p = base + (size_t)(row0 + m) * stride;
  const int ka = k0 + hi * 8, kb = ka + 16;
  v16h r;
  if (k0 + 32 <= KVALID) {               // compile-time foldable, no divergence
    const v4f a0 = ldg4(p + ka), a1 = ldg4(p + ka + 4);
    const v4f b0 = ldg4(p + kb), b1 = ldg4(p + kb + 4);
#pragma unroll
    for (int i = 0; i < 4; ++i) {
      r[i]      = (_Float16)a0[i];
      r[4 + i]  = (_Float16)a1[i];
      r[8 + i]  = (_Float16)b0[i];
      r[12 + i] = (_Float16)b1[i];
    }
  } else {                               // branch-free: clamp address + select
#pragma unroll
    for (int i = 0; i < 16; ++i) {
      const int k = (i < 8) ? (ka + i) : (kb + i - 8);
      const int kc = (k < KVALID - 1) ? k : (KVALID - 1);
      const float v = ldg1(p + kc);
      r[i] = (k < KVALID) ? (_Float16)v : (_Float16)0.f;
    }
  }
  return r;
}

// B fragment (32x16) from B^T in LDS (N x K row-major):
// lanes 0-15: K k0..k0+7, k0+16..23 ; lanes 16-31: K k0+8..15, k0+24..31 ; n = lane&15
template <int KVALID>
__device__ __forceinline__ v16h fragB(const _Float16* bt, int n0, int stride,
                                      int k0) {
  const int lane = threadIdx.x & 31;
  const int n = lane & 15, hi = lane >> 4;
  const _Float16* p = bt + (n0 + n) * stride + k0 + hi * 8;
  const v8h a = *(const v8h*)p;
  const v8h b = *(const v8h*)(p + 16);
  const int kbase = k0 + hi * 8;
  v16h r;
  if (k0 + 32 <= KVALID) {
#pragma unroll
    for (int i = 0; i < 8; ++i) { r[i] = a[i]; r[8 + i] = b[i]; }
  } else {
#pragma unroll
    for (int i = 0; i < 8; ++i) {
      r[i]     = (kbase + i < KVALID) ? a[i] : (_Float16)0.f;
      r[8 + i] = (kbase + 16 + i < KVALID) ? b[i] : (_Float16)0.f;
    }
  }
  return r;
}

// ---------------- geometry / LDS layout ----------------
constexpr int QSTR = 72, KSTR = 72, VTSTR = 136, PSTR = 136;
constexpr int OSTR = 68, XSTR = 104, YSTR = 64;
constexpr int WCAP = 64 * 168;

constexpr int F16TOT = 128 * QSTR + 128 * KSTR + 64 * VTSTR + 128 * PSTR +
                       WCAP + 2 * (128 * XSTR) + 2 * (128 * YSTR);
constexpr int F32TOT = 128 * OSTR + 128 + 256 * 16 + 16 + 544;
constexpr size_t SMEM_BYTES = (size_t)F16TOT * 2 + (size_t)F32TOT * 4;

struct KArgs {
  const float* in[42];
  float* out;
};

// ---------------- 128 x DOUT GEMM via WMMA ----------------
// MODE: 0 = f16 row-major, 1 = f16 transposed, 2 = f32 row-major
template <int MODE, int DOUT, int DIN, typename TA>
__device__ __forceinline__ void gemm(const TA* A, int astride,
                                     const _Float16* Bt, int bstride,
                                     const float* bias, _Float16* dst16,
                                     float* dst32, int dstride) {
  constexpr int KP = (DIN + 31) & ~31;
  constexpr int NTN = DOUT / 16;
  const int wave = threadIdx.x >> 5;
  const int lane = threadIdx.x & 31;
#pragma unroll
  for (int it = 0; it < NTN; ++it) {
    const int t = wave + it * 8;
    const int tm = t / NTN, tn = t % NTN;
    v8f c = {};
#pragma unroll
    for (int k0 = 0; k0 < KP; k0 += 32) {
      const v16h a = fragA<DIN>(A, tm * 16, astride, k0);
      const v16h b = fragB<DIN>(Bt, tn * 16, bstride, k0);
      c = wmma16(a, b, c);
    }
    const int col = tn * 16 + (lane & 15);
    const float bb = bias ? ldg1(bias + col) : 0.f;
#pragma unroll
    for (int r = 0; r < 8; ++r) {
      const int row = tm * 16 + r + ((lane >> 4) << 3);
      const float v = c[r] + bb;
      if (MODE == 2)      dst32[row * dstride + col] = v;
      else if (MODE == 1) dst16[col * dstride + row] = (_Float16)v;
      else                dst16[row * dstride + col] = (_Float16)v;
    }
  }
}

// W (DIN x DOUT row-major f32, global) -> W^T (DOUT x wstride f16, LDS)
template <int DIN, int DOUT>
__device__ __forceinline__ void load_wt(const float* Wg, int wstride,
                                        _Float16* sW) {
  for (int idx = threadIdx.x; idx < DIN * DOUT; idx += 256) {
    const int k = idx / DOUT, n = idx - k * DOUT;       // coalesced global read
    sW[n * wstride + k] = (_Float16)ldg1(Wg + idx);
  }
}

// ---------------- one cross-attention + layernorm (one side) ----------------
template <int DINQ, int DINKV, int D, typename TQ, typename TKV>
__device__ void attention(const TQ* Xq, int xqstride, const TKV* Xkv,
                          int xkvstride, const float* Wq, const float* bq,
                          const float* Wk, const float* bk, const float* Wv,
                          const float* bv, const float* lng, const float* lnb,
                          _Float16* Y,                        // LN out or null
                          const _Float16* res, int resstride, // residual or null
                          float* outF,                        // f32 out or null
                          _Float16* sQ, _Float16* sK, _Float16* sVt,
                          _Float16* sP, _Float16* sW, float* sO,
                          float* rowsum) {
  constexpr int WSQ = ((DINQ + 31) & ~31) + 8;
  constexpr int WSKV = ((DINKV + 31) & ~31) + 8;

  load_wt<DINQ, D>(Wq, WSQ, sW);
  __syncthreads();
  gemm<0, D, DINQ>(Xq, xqstride, sW, WSQ, bq, sQ, nullptr, QSTR);
  __syncthreads();
  load_wt<DINKV, D>(Wk, WSKV, sW);
  __syncthreads();
  gemm<0, D, DINKV>(Xkv, xkvstride, sW, WSKV, bk, sK, nullptr, KSTR);
  __syncthreads();
  load_wt<DINKV, D>(Wv, WSKV, sW);
  __syncthreads();
  gemm<1, D, DINKV>(Xkv, xkvstride, sW, WSKV, bv, sVt, nullptr, VTSTR);
  __syncthreads();
  // S = Q K^T (raw f16; scale folded into softmax)
  gemm<0, 128, D>(sQ, QSTR, sK, KSTR, nullptr, sP, nullptr, PSTR);
  __syncthreads();
  // softmax rows (unnormalized exp; 1/sum folded into LN pass)
  if (threadIdx.x < 128) {
    const int r = threadIdx.x;
    const float scale = rsqrtf((float)D);
    _Float16* prow = sP + r * PSTR;
    float mx = -3.4e38f;
#pragma unroll
    for (int j0 = 0; j0 < 128; j0 += 8) {
      const v8h v = *(const v8h*)(prow + j0);
#pragma unroll
      for (int i = 0; i < 8; ++i) mx = fmaxf(mx, (float)v[i]);
    }
    float sum = 0.f;
#pragma unroll
    for (int j0 = 0; j0 < 128; j0 += 8) {
      v8h v = *(const v8h*)(prow + j0);
      v8h e16;
#pragma unroll
      for (int i = 0; i < 8; ++i) {
        const float e = __expf(((float)v[i] - mx) * scale);
        sum += e;
        e16[i] = (_Float16)e;
      }
      *(v8h*)(prow + j0) = e16;
    }
    rowsum[r] = sum;
  }
  __syncthreads();
  // O = P @ V (f32)
  gemm<2, D, 128>(sP, PSTR, sVt, VTSTR, nullptr, nullptr, sO, OSTR);
  __syncthreads();
  // layernorm (+residual)
  if (threadIdx.x < 128) {
    const int r = threadIdx.x;
    const float inv = 1.f / rowsum[r];
    const float* orow = sO + r * OSTR;
    float s1 = 0.f, s2 = 0.f;
#pragma unroll
    for (int j = 0; j < D; ++j) {
      const float v = orow[j] * inv;
      s1 += v;
      s2 += v * v;
    }
    const float mean = s1 / D;
    const float var = s2 / D - mean * mean;
    const float rstd = rsqrtf(var + 1e-5f);
#pragma unroll
    for (int j = 0; j < D; ++j) {
      float y = (orow[j] * inv - mean) * rstd * ldg1(lng + j) + ldg1(lnb + j);
      if (res) y += (float)res[r * resstride + j];
      if (Y) Y[r * YSTR + j] = (_Float16)y;
      if (outF) outF[r * 16 + j] = y;
    }
  }
  __syncthreads();
}

template <int D>
__device__ __forceinline__ void commit(const _Float16* Y, _Float16* X) {
  constexpr int G = D / 8;
  for (int idx = threadIdx.x; idx < 128 * G; idx += 256) {
    const int r = idx / G, j0 = (idx - r * G) * 8;
    *(v8h*)(X + r * XSTR + 8 + j0) = *(const v8h*)(Y + r * YSTR + j0);
  }
}

// ---------------- fused kernel: one block per (b,c) ----------------
__launch_bounds__(256, 1)
__global__ void soap_fused_kernel(KArgs args) {
  extern __shared__ char smem[];
  _Float16* sQ  = (_Float16*)smem;
  _Float16* sK  = sQ  + 128 * QSTR;
  _Float16* sVt = sK  + 128 * KSTR;
  _Float16* sP  = sVt + 64 * VTSTR;
  _Float16* sW  = sP  + 128 * PSTR;
  _Float16* sXL = sW  + WCAP;
  _Float16* sXH = sXL + 128 * XSTR;
  _Float16* sYL = sXH + 128 * XSTR;
  _Float16* sYH = sYL + 128 * YSTR;
  float* sO     = (float*)(sYH + 128 * YSTR);
  float* rowsum = sO + 128 * OSTR;
  float* sOut   = rowsum + 128;
  float* accum  = sOut + 256 * 16;
  float* aggW   = accum + 16;

  const int bc = blockIdx.x;
  const float* nLnce = args.in[0] + (size_t)bc * 128 * 72;
  const float* nHnce = args.in[1] + (size_t)bc * 128 * 72;
  const float* nLenv = args.in[2] + (size_t)bc * 128 * 136;
  const float* nHenv = args.in[3] + (size_t)bc * 128 * 136;
  const float* nLg   = args.in[4] + (size_t)bc * 128 * 8;
  const float* nHg   = args.in[5] + (size_t)bc * 128 * 8;

  __builtin_prefetch(nLnce, 0, 0);
  __builtin_prefetch(nHenv, 0, 0);

  // x{L,H} cols 0..7 = raw node features (persist across stages)
  for (int idx = threadIdx.x; idx < 128; idx += 256) {
    const v4f a0 = ldg4(nLg + idx * 8), a1 = ldg4(nLg + idx * 8 + 4);
    const v4f b0 = ldg4(nHg + idx * 8), b1 = ldg4(nHg + idx * 8 + 4);
    v8h l, h;
#pragma unroll
    for (int i = 0; i < 4; ++i) {
      l[i] = (_Float16)a0[i]; l[4 + i] = (_Float16)a1[i];
      h[i] = (_Float16)b0[i]; h[4 + i] = (_Float16)b1[i];
    }
    *(v8h*)(sXL + idx * XSTR) = l;
    *(v8h*)(sXH + idx * XSTR) = h;
  }
  __syncthreads();

  const float* const* P = args.in;

  // ---- stage 1: q from *_nce, kv from other-class env; d=64 ----
  attention<72, 136, 64>(nLnce, 72, nHenv, 136,
                         P[6], P[7], P[8], P[9], P[10], P[11], P[12], P[13],
                         sYL, (const _Float16*)nullptr, 0, (float*)nullptr,
                         sQ, sK, sVt, sP, sW, sO, rowsum);
  attention<72, 136, 64>(nHnce, 72, nLenv, 136,
                         P[6], P[7], P[8], P[9], P[10], P[11], P[12], P[13],
                         sYH, (const _Float16*)nullptr, 0, (float*)nullptr,
                         sQ, sK, sVt, sP, sW, sO, rowsum);
  commit<64>(sYL, sXL);
  commit<64>(sYH, sXH);
  __syncthreads();

  // ---- stage 2: din=72, d=32 ----
  attention<72, 72, 32>(sXL, XSTR, sXH, XSTR,
                        P[14], P[15], P[16], P[17], P[18], P[19], P[20], P[21],
                        sYL, (const _Float16*)nullptr, 0, (float*)nullptr,
                        sQ, sK, sVt, sP, sW, sO, rowsum);
  attention<72, 72, 32>(sXH, XSTR, sXL, XSTR,
                        P[14], P[15], P[16], P[17], P[18], P[19], P[20], P[21],
                        sYH, (const _Float16*)nullptr, 0, (float*)nullptr,
                        sQ, sK, sVt, sP, sW, sO, rowsum);
  commit<32>(sYL, sXL);
  commit<32>(sYH, sXH);
  __syncthreads();

  // ---- stage 3: din=40, d=16 (o3 lives on as x cols 8..23) ----
  attention<40, 40, 16>(sXL, XSTR, sXH, XSTR,
                        P[22], P[23], P[24], P[25], P[26], P[27], P[28], P[29],
                        sYL, (const _Float16*)nullptr, 0, (float*)nullptr,
                        sQ, sK, sVt, sP, sW, sO, rowsum);
  attention<40, 40, 16>(sXH, XSTR, sXL, XSTR,
                        P[22], P[23], P[24], P[25], P[26], P[27], P[28], P[29],
                        sYH, (const _Float16*)nullptr, 0, (float*)nullptr,
                        sQ, sK, sVt, sP, sW, sO, rowsum);
  commit<16>(sYL, sXL);
  commit<16>(sYH, sXH);
  __syncthreads();

  // ---- stage 4: din=24, d=16; out = LN(attn) + o3 (residual = x cols 8..23) ----
  attention<24, 24, 16>(sXL, XSTR, sXH, XSTR,
                        P[30], P[31], P[32], P[33], P[34], P[35], P[36], P[37],
                        (_Float16*)nullptr, sXL + 8, XSTR, sOut,
                        sQ, sK, sVt, sP, sW, sO, rowsum);
  attention<24, 24, 16>(sXH, XSTR, sXL, XSTR,
                        P[30], P[31], P[32], P[33], P[34], P[35], P[36], P[37],
                        (_Float16*)nullptr, sXH + 8, XSTR, sOut + 128 * 16,
                        sQ, sK, sVt, sP, sW, sO, rowsum);

  // ---- aggregator: relu(out@W1+b1) summed over 256 atoms, then @W2+b2 ----
  // aggW layout: W1[0..255], b1[256..271], W2[272..527], b2[528..543]
  {
    const int i = threadIdx.x;
    aggW[i]       = ldg1(P[38] + i);
    aggW[272 + i] = ldg1(P[40] + i);
  }
  if (threadIdx.x < 16) {
    aggW[256 + threadIdx.x] = ldg1(P[39] + threadIdx.x);
    aggW[528 + threadIdx.x] = ldg1(P[41] + threadIdx.x);
    accum[threadIdx.x] = 0.f;
  }
  __syncthreads();
  {
    const float* x = sOut + threadIdx.x * 16;  // one atom per thread
#pragma unroll
    for (int j = 0; j < 16; ++j) {
      float s = aggW[256 + j];
#pragma unroll
      for (int i = 0; i < 16; ++i) s += x[i] * aggW[i * 16 + j];
      atomicAdd(&accum[j], fmaxf(s, 0.f));
    }
  }
  __syncthreads();
  if (threadIdx.x < 16) {
    const int j = threadIdx.x;
    float s = aggW[528 + j];
#pragma unroll
    for (int i = 0; i < 16; ++i) s += accum[i] * aggW[272 + i * 16 + j];
    stg1(args.out + (size_t)bc * 16 + j, s);
  }
}

extern "C" void kernel_launch(void* const* d_in, const int* in_sizes, int n_in,
                              void* d_out, int out_size, void* d_ws, size_t ws_size,
                              hipStream_t stream) {
  (void)in_sizes; (void)out_size; (void)d_ws; (void)ws_size;
  KArgs a;
  const int n = (n_in < 42) ? n_in : 42;
  for (int i = 0; i < n; ++i) a.in[i] = (const float*)d_in[i];
  for (int i = n; i < 42; ++i) a.in[i] = nullptr;
  a.out = (float*)d_out;

  (void)hipFuncSetAttribute((const void*)soap_fused_kernel,
                            hipFuncAttributeMaxDynamicSharedMemorySize,
                            (int)SMEM_BYTES);
  soap_fused_kernel<<<8 * 256, 256, SMEM_BYTES, stream>>>(a);
}
